// PGSeg_37847251812716
// MI455X (gfx1250) — matrix operand order repaired and checked by reading, not compiled, not run on hardware
//
#include <hip/hip_runtime.h>
#include <hip/hip_bf16.h>
#include <stdint.h>

// ---------------------------------------------------------------------------
// PGSeg forward on gfx1250 (MI455X), wave32 + WMMA bf16 + async-LDS + TDM.
// Forward of hard_softmax straight-through == y_hard, so the pipeline is:
//   bf16 convert -> WMMA GEMMs (q,k,v) with double-buffered async-LDS tiles
//   -> WMMA scores + argmax -> segment mean via LDS float atomics -> WMMA
//   output projection.
// ---------------------------------------------------------------------------

typedef unsigned short u16;
typedef unsigned int   u32;
typedef __attribute__((ext_vector_type(16))) __bf16 v16bf;
typedef __attribute__((ext_vector_type(8)))  float  v8f;
typedef __attribute__((ext_vector_type(4)))  unsigned int v4u;
typedef __attribute__((ext_vector_type(8)))  int    v8i;
typedef __attribute__((ext_vector_type(4)))  int    v4i;

union FragAB {
  v16bf v;
  uint4 q[2];
};

__device__ __forceinline__ u16 f32_to_bf16(float f) {
  union { float f; unsigned int u; } x; x.f = f;
  unsigned int u = x.u;
  u += 0x7FFFu + ((u >> 16) & 1u);   // round-to-nearest-even
  return (u16)(u >> 16);
}
__device__ __forceinline__ float bf16_to_f32(u16 h) {
  union { unsigned int u; float f; } x; x.u = ((unsigned int)h) << 16;
  return x.f;
}

// Async global -> LDS copy, 16B per lane, tracked with ASYNCcnt.
// the LDS destination address comes from the vdst VGPR; generic LDS pointers carry the LDS byte
// offset in their low 32 bits.
__device__ __forceinline__ void async_load_b128(void* lds_ptr, const void* g) {
  unsigned lds_off = (unsigned)(size_t)lds_ptr;
  asm volatile("global_load_async_to_lds_b128 %0, %1, off"
               :: "v"(lds_off), "v"(g) : "memory");
}
__device__ __forceinline__ void wait_async0() {
  asm volatile("s_wait_asynccnt 0" ::: "memory");
}

// --------------------------- fp32 -> bf16 ----------------------------------
__global__ __launch_bounds__(256) void cvt_bf16_kernel(const float* __restrict__ in,
                                                       u16* __restrict__ out, int n) {
  int i = blockIdx.x * 256 + threadIdx.x;
  if (i < n) out[i] = f32_to_bf16(in[i]);
}

// --------------------------- zero workspace --------------------------------
__global__ __launch_bounds__(256) void zero_kernel(float* __restrict__ acc, int nacc,
                                                   int* __restrict__ cnt, int ncnt) {
  int i = blockIdx.x * 256 + threadIdx.x;
  if (i < nacc) acc[i] = 0.0f;
  if (i < ncnt) cnt[i] = 0;
}

// --------------------------- tiled bf16 GEMM -------------------------------
// Y[M,N] = X[M,K] @ W[N,K]^T + bias      (torch Linear convention)
// Block tile 64(M) x 128(N) x 32(K), 8 waves in a 2x4 grid, each wave 32x32.
// Double-buffered LDS tiles staged with GLOBAL_LOAD_ASYNC_TO_LDS_B128; next
// tile's async loads are issued right after the barrier and only waited at
// the top of the next iteration, overlapping DMA with WMMA.
// WMMA fragment layouts per CDNA5 ISA 7.12.2:
//   A (16x32 bf16): row = lane%16; elem e -> K = (e/8)*16 + (lane/16)*8 + e%8
//   B (32x16 bf16): col = lane%16; elem e -> K = (e/8)*16 + (lane/16)*8 + e%8
//   D (16x16 f32) : col = lane%16; vgpr r -> M = r + 8*(lane/16)
#define TILE_M 64
#define TILE_N 128
#define TILE_K 32

__global__ __launch_bounds__(256)
void gemm_bf16_kernel(const u16* __restrict__ X, const u16* __restrict__ W,
                      const float* __restrict__ bias, void* __restrict__ Y,
                      int M, int N, int K, int out_is_bf16) {
  __shared__ u16 sA[2][TILE_M][TILE_K];   // 8 KB
  __shared__ u16 sB[2][TILE_N][TILE_K];   // 16 KB

  const int t    = threadIdx.x;
  const int lane = t & 31;
  const int wv   = t >> 5;
  const int wm   = wv & 1;          // 2 wave rows (M)
  const int wn   = wv >> 1;         // 4 wave cols (N)
  const int hi   = lane >> 4;
  const int l16  = lane & 15;
  const int m0   = blockIdx.y * TILE_M;
  const int n0   = blockIdx.x * TILE_N;

  // Stage one 64x32 A tile + 128x32 B tile into LDS buffer `buf` (async).
  auto stage = [&](int k0, int buf) {
    {
      int r = t >> 2, qi = t & 3;
      async_load_b128(&sA[buf][r][qi * 8],
                      &X[(size_t)(m0 + r) * K + k0 + qi * 8]);
    }
    for (int i = 0; i < 2; ++i) {
      int idx = t + i * 256;
      int r = idx >> 2, qi = idx & 3;
      async_load_b128(&sB[buf][r][qi * 8],
                      &W[(size_t)(n0 + r) * K + k0 + qi * 8]);
    }
  };

  v8f acc[2][2] = {};

  const int nsteps = K / TILE_K;
  stage(0, 0);
  for (int ks = 0; ks < nsteps; ++ks) {
    const int buf = ks & 1;
    wait_async0();        // my async writes into `buf` are done
    __syncthreads();      // everyone's are done
    if (ks + 1 < nsteps) stage((ks + 1) * TILE_K, buf ^ 1);

    FragAB bfrag[2];
    for (int nt = 0; nt < 2; ++nt) {
      int col = wn * 32 + nt * 16 + l16;
      bfrag[nt].q[0] = *reinterpret_cast<const uint4*>(&sB[buf][col][hi * 8]);
      bfrag[nt].q[1] = *reinterpret_cast<const uint4*>(&sB[buf][col][16 + hi * 8]);
    }
    for (int mt = 0; mt < 2; ++mt) {
      int row = wm * 32 + mt * 16 + l16;
      FragAB afrag;
      afrag.q[0] = *reinterpret_cast<const uint4*>(&sA[buf][row][hi * 8]);
      afrag.q[1] = *reinterpret_cast<const uint4*>(&sA[buf][row][16 + hi * 8]);
      for (int nt = 0; nt < 2; ++nt) {
        acc[mt][nt] = __builtin_amdgcn_wmma_f32_16x16x32_bf16(
            false, afrag.v, false, bfrag[nt].v, (short)0, acc[mt][nt], false, false);
      }
    }
    // No trailing barrier needed: the s_wait_dscnt before the WMMAs orders
    // each wave's LDS reads before it signals the next iteration's barrier.
  }

  // Epilogue: bias add + store (bf16 or f32).
  for (int mt = 0; mt < 2; ++mt) {
    for (int nt = 0; nt < 2; ++nt) {
      int colg = n0 + wn * 32 + nt * 16 + l16;
      float bv = bias[colg];
      for (int r = 0; r < 8; ++r) {
        int rowg = m0 + wm * 32 + mt * 16 + hi * 8 + r;
        float val = acc[mt][nt][r] + bv;
        if (out_is_bf16)
          reinterpret_cast<u16*>(Y)[(size_t)rowg * N + colg] = f32_to_bf16(val);
        else
          reinterpret_cast<float*>(Y)[(size_t)rowg * N + colg] = val;
      }
    }
  }
}

// ------------------- scores + argmax + segment accumulate ------------------
// One block per (b, h, 256-wide S chunk). Q head tile (64x96 bf16) pulled
// into LDS by the Tensor Data Mover (one tensor_load_to_lds per block,
// TENSORcnt-tracked), K fragments straight from global, 64x16 score strips
// via 12 WMMAs, argmax over N=64 in registers + __shfl_xor(16), segment-sum
// of V into LDS via ds_add_f32, then one atomic flush per block.
__global__ __launch_bounds__(256)
void score_assign_kernel(const u16* __restrict__ qp, const u16* __restrict__ kp,
                         const u16* __restrict__ vp, float* __restrict__ g_accum,
                         int* __restrict__ g_count) {
  __shared__ u16   sQ[64][96];     // 12 KB
  __shared__ float sAcc[64][96];   // 24 KB
  __shared__ int   sCnt[64];

  const int t    = threadIdx.x;
  const int lane = t & 31;
  const int wv   = t >> 5;
  const int hi   = lane >> 4;
  const int l16  = lane & 15;
  const int b    = blockIdx.z;
  const int h    = blockIdx.y;
  const int s_chunk = blockIdx.x * 256;

  // Zero LDS accumulators.
  for (int i = t; i < 64 * 96; i += 256) (&sAcc[0][0])[i] = 0.0f;
  if (t < 64) sCnt[t] = 0;

  // Q head tile: rows n=0..63 (stride 768 bf16), 96 bf16 each, starting at
  // column h*96 of batch b.
#if __has_builtin(__builtin_amdgcn_tensor_load_to_lds)
  if (wv == 0) {
    // Tensor DMA descriptor (D#): 2D tile, data_size=2B, tile 96x64,
    // tensor_dim0=96, tensor_dim1=64, tensor_dim0_stride=768.
    size_t ga = (size_t)qp + ((size_t)(b * 64) * 768 + h * 96) * 2;
    v4u g0 = { 1u,                                    // count=1, user mode
               (u32)(size_t)&sQ[0][0],                // lds_addr
               (u32)ga,                                // global_addr[31:0]
               (u32)((ga >> 32) & 0x01FFFFFFu) | 0x80000000u };  // type=2
    v8i g1 = { (int)0x00010000,         // data_size=1 (2 bytes)
               (int)(96u << 16),        // tensor_dim0 = 96 (bits 79:48)
               (int)(64u << 16),        // tensor_dim1 = 64 (bits 111:80)
               (int)(96u << 16),        // tile_dim0 = 96   (bits 127:112)
               64,                      // tile_dim1 = 64   (bits 143:128)
               768,                     // tensor_dim0_stride (bits 207:160)
               0, 0 };
    v4i gz4 = { 0, 0, 0, 0 };
    v8i gz8 = { 0, 0, 0, 0, 0, 0, 0, 0 };
    __builtin_amdgcn_tensor_load_to_lds(g0, g1, gz4, gz4, gz8, 0);
    __builtin_amdgcn_s_wait_tensorcnt(0);
  }
#else
  for (int i = t; i < 768; i += 256) {
    int n = i / 12, j = i % 12;
    async_load_b128(&sQ[n][j * 8],
                    &qp[((size_t)(b * 64 + n)) * 768 + h * 96 + j * 8]);
  }
  wait_async0();
#endif
  __syncthreads();

  for (int strip = 0; strip < 2; ++strip) {
    const int s_base = s_chunk + (wv * 2 + strip) * 16;

    v8f acc4[4] = {};
    for (int kk = 0; kk < 96; kk += 32) {
      FragAB bf;
      {
        int s = s_base + l16;
        const u16* kr = kp + ((size_t)(b * 4096 + s)) * 768 + h * 96 + kk + hi * 8;
        bf.q[0] = *reinterpret_cast<const uint4*>(kr);
        bf.q[1] = *reinterpret_cast<const uint4*>(kr + 16);
      }
      for (int mt = 0; mt < 4; ++mt) {
        int row = mt * 16 + l16;
        FragAB af;
        af.q[0] = *reinterpret_cast<const uint4*>(&sQ[row][kk + hi * 8]);
        af.q[1] = *reinterpret_cast<const uint4*>(&sQ[row][kk + 16 + hi * 8]);
        acc4[mt] = __builtin_amdgcn_wmma_f32_16x16x32_bf16(
            false, af.v, false, bf.v, (short)0, acc4[mt], false, false);
      }
    }

    // Argmax over the 64 query slots for this lane's column (scale>0 is
    // argmax-invariant; softmax forward is y_hard). Ascending n order inside
    // the loop + strict '>' gives first-max semantics like jnp.argmax.
    float bestv = -3.4e38f;
    int   bestn = 0;
    for (int mt = 0; mt < 4; ++mt)
      for (int r = 0; r < 8; ++r) {
        float vv = acc4[mt][r];
        int n = mt * 16 + hi * 8 + r;
        if (vv > bestv) { bestv = vv; bestn = n; }
      }
    float ov = __shfl_xor(bestv, 16);
    int   on = __shfl_xor(bestn, 16);
    if (ov > bestv || (ov == bestv && on < bestn)) { bestv = ov; bestn = on; }

    if (lane < 16) atomicAdd(&sCnt[bestn], 1);

    // Segment-sum assigned V rows into LDS (ds_add_f32).
    for (int j = 0; j < 16; ++j) {
      int nsel = __shfl(bestn, j);
      int s = s_base + j;
      const u16* vr = vp + ((size_t)(b * 4096 + s)) * 768 + h * 96;
      for (int c = lane; c < 96; c += 32)
        atomicAdd(&sAcc[nsel][c], bf16_to_f32(vr[c]));
    }
  }
  __syncthreads();

  // Flush block-local partials to global accumulators.
  float* ga = g_accum + ((size_t)(b * 8 + h)) * 64 * 96;
  for (int i = t; i < 64 * 96; i += 256) atomicAdd(&ga[i], (&sAcc[0][0])[i]);
  if (t < 64) atomicAdd(&g_count[(b * 8 + h) * 64 + t], sCnt[t]);
}

// --------------- normalize by (count + eps) and merge heads ----------------
__global__ __launch_bounds__(256)
void finalize_kernel(const float* __restrict__ acc, const int* __restrict__ cnt,
                     u16* __restrict__ x2) {
  int i = blockIdx.x * 256 + threadIdx.x;   // over B*N*C = 512*768
  if (i >= 512 * 768) return;
  int cf = i % 768, bn = i / 768;
  int n = bn % 64, b = bn / 64;
  int h = cf / 96, c = cf % 96;
  float denom = (float)cnt[(b * 8 + h) * 64 + n] + 1.0f;   // ASSIGN_EPS = 1.0
  float v = acc[(((size_t)(b * 8 + h) * 64) + n) * 96 + c] / denom;
  x2[i] = f32_to_bf16(v);
}

// ---------------------------------------------------------------------------
extern "C" void kernel_launch(void* const* d_in, const int* in_sizes, int n_in,
                              void* d_out, int out_size, void* d_ws, size_t ws_size,
                              hipStream_t stream) {
  (void)in_sizes; (void)n_in; (void)out_size; (void)ws_size;
  const int B = 8, N = 64, S = 4096, C = 768, H = 8;
  const int BN = B * N;        // 512
  const int BS = B * S;        // 32768

  const float* query = (const float*)d_in[0];
  const float* key   = (const float*)d_in[1];
  const float* Wq = (const float*)d_in[2];  const float* bq = (const float*)d_in[3];
  const float* Wk = (const float*)d_in[4];  const float* bk = (const float*)d_in[5];
  const float* Wv = (const float*)d_in[6];  const float* bv = (const float*)d_in[7];
  const float* Wp = (const float*)d_in[8];  const float* bp = (const float*)d_in[9];
  float* out = (float*)d_out;

  char* ws = (char*)d_ws;
  size_t off = 0;
  auto alloc = [&](size_t bytes) {
    off = (off + 255) & ~(size_t)255;
    void* p = ws + off;
    off += bytes;
    return p;
  };

  u16* qb   = (u16*)alloc((size_t)BN * C * 2);
  u16* keyb = (u16*)alloc((size_t)BS * C * 2);
  u16* wqb  = (u16*)alloc((size_t)C * C * 2);
  u16* wkb  = (u16*)alloc((size_t)C * C * 2);
  u16* wvb  = (u16*)alloc((size_t)C * C * 2);
  u16* wpb  = (u16*)alloc((size_t)C * C * 2);
  u16* qp   = (u16*)alloc((size_t)BN * C * 2);
  u16* kp   = (u16*)alloc((size_t)BS * C * 2);
  u16* vp   = (u16*)alloc((size_t)BS * C * 2);
  float* g_accum = (float*)alloc((size_t)B * H * 64 * 96 * 4);
  int*   g_count = (int*)alloc((size_t)B * H * 64 * 4);
  u16* x2   = (u16*)alloc((size_t)BN * C * 2);

  // 1) fp32 -> bf16 conversions.
  int nq = BN * C, nk = BS * C, nw = C * C;
  cvt_bf16_kernel<<<(nq + 255) / 256, 256, 0, stream>>>(query, qb, nq);
  cvt_bf16_kernel<<<(nk + 255) / 256, 256, 0, stream>>>(key, keyb, nk);
  cvt_bf16_kernel<<<(nw + 255) / 256, 256, 0, stream>>>(Wq, wqb, nw);
  cvt_bf16_kernel<<<(nw + 255) / 256, 256, 0, stream>>>(Wk, wkb, nw);
  cvt_bf16_kernel<<<(nw + 255) / 256, 256, 0, stream>>>(Wv, wvb, nw);
  cvt_bf16_kernel<<<(nw + 255) / 256, 256, 0, stream>>>(Wp, wpb, nw);

  // 2) q/k/v projections via WMMA GEMM (y = x @ W.T + b).
  dim3 gq(C / TILE_N, BN / TILE_M);     // 6 x 8
  gemm_bf16_kernel<<<gq, 256, 0, stream>>>(qb, wqb, bq, qp, BN, C, C, 1);
  dim3 gkv(C / TILE_N, BS / TILE_M);    // 6 x 512
  gemm_bf16_kernel<<<gkv, 256, 0, stream>>>(keyb, wkb, bk, kp, BS, C, C, 1);
  gemm_bf16_kernel<<<gkv, 256, 0, stream>>>(keyb, wvb, bv, vp, BS, C, C, 1);

  // 3) zero segment accumulators.
  int nacc = B * H * 64 * 96, ncnt = B * H * 64;
  zero_kernel<<<(nacc + 255) / 256, 256, 0, stream>>>(g_accum, nacc, g_count, ncnt);

  // 4) scores + argmax + segment accumulate.
  dim3 gsc(S / 256, H, B);              // 16 x 8 x 8
  score_assign_kernel<<<gsc, 256, 0, stream>>>(qp, kp, vp, g_accum, g_count);

  // 5) normalize by (count + 1) and merge heads into [B,N,C] bf16.
  int nfin = BN * C;
  finalize_kernel<<<(nfin + 255) / 256, 256, 0, stream>>>(g_accum, g_count, x2);

  // 6) output projection -> fp32 result.
  dim3 gp(C / TILE_N, BN / TILE_M);
  gemm_bf16_kernel<<<gp, 256, 0, stream>>>(x2, wpb, bp, out, BN, C, C, 0);
}
